// STGAT_72181220376901
// MI455X (gfx1250) — matrix operand order repaired and checked by reading, not compiled, not run on hardware
//
#include <hip/hip_runtime.h>
#include <hip/hip_bf16.h>
#include <math.h>

typedef __attribute__((ext_vector_type(16))) _Float16 v16h;
typedef __attribute__((ext_vector_type(8)))  _Float16 v8h;
typedef __attribute__((ext_vector_type(8)))  float    v8f;

#define NCOL 128

// pack two f32 -> packed f16 pair (one LDS dword)
__device__ __forceinline__ unsigned pack2h(float a, float b) {
  union { _Float16 h[2]; unsigned u; } p;
  p.h[0] = (_Float16)a;
  p.h[1] = (_Float16)b;
  return p.u;
}

// ---------------------------------------------------------------------------
// Generic WMMA GEMM: C[M,128] = actA(A[M,K]) @ B[K,128] (+ bias)
// block = 256 threads (8 waves), each block computes a 128x128 tile.
// Wave w computes rows [w*16, w*16+16) over all 128 columns (8 wmma tiles).
// A staged f32->f16 (optional elu) with packed b32 LDS stores; B staged
// transposed to [n][k] (k-pairs packed) so fragments are contiguous v8h loads.
// ---------------------------------------------------------------------------
__global__ __launch_bounds__(256)
void gemm128_wmma(const float* __restrict__ A, const float* __restrict__ B,
                  const float* __restrict__ bias, float* __restrict__ C,
                  int K, int applyElu) {
  __shared__ alignas(16) _Float16 lsA[128][40];
  __shared__ alignas(16) _Float16 lsB[128][40];
  const int tid  = threadIdx.x;
  const int wid  = tid >> 5;
  const int lane = tid & 31;
  const int m0   = blockIdx.x * 128;

  v8f zero = {};
  v8f acc[8];
#pragma unroll
  for (int j = 0; j < 8; ++j) acc[j] = zero;

  for (int k0 = 0; k0 < K; k0 += 32) {
    // ---- stage A tile (128 x 32): 16 consecutive f32 per thread -> 8 dwords
    {
      const int r  = tid >> 1;
      const int cb = (tid & 1) * 16;
      const float* ap = A + (size_t)(m0 + r) * K + k0 + cb;
      unsigned* dp = (unsigned*)&lsA[r][cb];
#pragma unroll
      for (int p = 0; p < 8; ++p) {
        float v0 = ap[2 * p], v1 = ap[2 * p + 1];
        if (applyElu) {
          v0 = (v0 > 0.f) ? v0 : (expf(v0) - 1.f);
          v1 = (v1 > 0.f) ? v1 : (expf(v1) - 1.f);
        }
        dp[p] = pack2h(v0, v1);
      }
    }
    // ---- stage B tile (32 x 128) transposed -> lsB[n][k], k-pairs packed ----
    {
      const int kp = tid >> 4;        // 16 k-pairs
      const int n0 = (tid & 15) * 8;  // 8 columns per thread
      const float* b0 = B + (size_t)(k0 + 2 * kp) * NCOL;
      const float* b1 = b0 + NCOL;
#pragma unroll
      for (int i = 0; i < 8; ++i) {
        const int n = n0 + i;
        *(unsigned*)&lsB[n][2 * kp] = pack2h(b0[n], b1[n]);
      }
    }
    __syncthreads();

    // ---- A fragment: lane l holds row (l&15), K = kh..kh+7 and kh+16..kh+23
    const int ar  = wid * 16 + (lane & 15);
    const int akh = (lane >> 4) * 8;
    union { v16h v; v8h h[2]; } ua;
    ua.h[0] = *(const v8h*)&lsA[ar][akh];
    ua.h[1] = *(const v8h*)&lsA[ar][akh + 16];

#pragma unroll
    for (int j = 0; j < 8; ++j) {
      const int bn  = j * 16 + (lane & 15);
      const int bkh = (lane >> 4) * 8;
      union { v16h v; v8h h[2]; } ub;
      ub.h[0] = *(const v8h*)&lsB[bn][bkh];
      ub.h[1] = *(const v8h*)&lsB[bn][bkh + 16];
      acc[j] = __builtin_amdgcn_wmma_f32_16x16x32_f16(
          false, ua.v, false, ub.v, (short)0, acc[j], false, false);
    }
    __syncthreads();
  }

  // ---- epilogue: C layout VGPR r <-> M = r + 8*(lane>=16), col = lane&15
  const int mrow = m0 + wid * 16 + (lane >> 4) * 8;
  const int col0 = lane & 15;
#pragma unroll
  for (int j = 0; j < 8; ++j) {
    const int c  = j * 16 + col0;
    const float bv = bias ? bias[c] : 0.f;
#pragma unroll
    for (int r = 0; r < 8; ++r)
      C[(size_t)(mrow + r) * NCOL + c] = acc[j][r] + bv;
  }
}

// ---------------------------------------------------------------------------
// Per-node attention dots + self-loop softmax-denominator init.
// a_src[n,h] = <h[n,h,:], att_src[h,:]>, z[n,h] = exp(lrelu(a_src+a_dst)).
// ---------------------------------------------------------------------------
__global__ __launch_bounds__(256)
void attn_node(const float* __restrict__ h, const float* __restrict__ att_s,
               const float* __restrict__ att_d, float* __restrict__ as,
               float* __restrict__ ad, float* __restrict__ z, int N, int H) {
  __shared__ float ss[128], sd[128];
  if (threadIdx.x < 128) {
    ss[threadIdx.x] = att_s[threadIdx.x];
    sd[threadIdx.x] = att_d[threadIdx.x];
  }
  __syncthreads();
  const int n = blockIdx.x * blockDim.x + threadIdx.x;
  if (n >= N) return;
  const int C = 128 / H;
  for (int hh = 0; hh < H; ++hh) {
    float a = 0.f, b = 0.f;
    for (int c = 0; c < C; ++c) {
      const float v = h[(size_t)n * 128 + hh * C + c];
      a += v * ss[hh * C + c];
      b += v * sd[hh * C + c];
    }
    as[n * H + hh] = a;
    ad[n * H + hh] = b;
    float l = a + b;
    l = (l > 0.f) ? l : 0.2f * l;
    z[n * H + hh] = expf(l);  // self-loop term (no max-sub; logits are O(1))
  }
}

// ---- softmax denominator over edges (thread per edge) ----
__global__ __launch_bounds__(256)
void edge_denom(const int* __restrict__ src, const int* __restrict__ dst,
                const float* __restrict__ as, const float* __restrict__ ad,
                float* __restrict__ z, int E, int H) {
  const int e = blockIdx.x * blockDim.x + threadIdx.x;
  if (e >= E) return;
  const int s = src[e], d = dst[e];
  for (int hh = 0; hh < H; ++hh) {
    float l = as[s * H + hh] + ad[d * H + hh];
    l = (l > 0.f) ? l : 0.2f * l;
    atomicAdd(&z[d * H + hh], expf(l));
  }
}

// ---- aggregation init: out = bias (+extra) + alpha_self * h ----
__global__ __launch_bounds__(256)
void agg_init(const float* __restrict__ hsrc, const float* __restrict__ bias,
              const float* __restrict__ extra, const float* __restrict__ as,
              const float* __restrict__ ad, const float* __restrict__ z,
              float* __restrict__ out, int N, int H) {
  const int i = blockIdx.x * blockDim.x + threadIdx.x;
  if (i >= N * 128) return;
  const int n = i >> 7, c = i & 127;
  const int hh = (c * H) >> 7;
  float l = as[n * H + hh] + ad[n * H + hh];
  l = (l > 0.f) ? l : 0.2f * l;
  const float alpha = expf(l) / (z[n * H + hh] + 1e-16f);
  float v = bias[c] + alpha * hsrc[i];
  if (extra) v += extra[i];
  out[i] = v;
}

// ---- message aggregation: wave per edge, 4 channels per lane ----
__global__ __launch_bounds__(256)
void edge_agg(const int* __restrict__ src, const int* __restrict__ dst,
              const float* __restrict__ h, const float* __restrict__ as,
              const float* __restrict__ ad, const float* __restrict__ z,
              float* __restrict__ out, int E, int H) {
  const int e = blockIdx.x * 8 + (threadIdx.x >> 5);
  const int lane = threadIdx.x & 31;
  if (e >= E) return;
  const int s = src[e], d = dst[e];
  float alpha[2];
  for (int hh = 0; hh < H; ++hh) {
    float l = as[s * H + hh] + ad[d * H + hh];
    l = (l > 0.f) ? l : 0.2f * l;
    alpha[hh] = expf(l) / (z[d * H + hh] + 1e-16f);
  }
#pragma unroll
  for (int q = 0; q < 4; ++q) {
    const int idx = lane + q * 32;
    const int hh  = (idx * H) >> 7;
    atomicAdd(&out[(size_t)d * 128 + idx], alpha[hh] * h[(size_t)s * 128 + idx]);
  }
}

// ---------------------------------------------------------------------------
// TCN (3-tap 'SAME' conv along T=4 within each block of 4 rows) as three
// shifted/masked WMMA GEMM accumulations, fused with bias + LayerNorm.
// tcn_w is [OUT=128][IN=128][3] (OIH); staged transposed per tap.
// ---------------------------------------------------------------------------
__global__ __launch_bounds__(256)
void tcn_ln_wmma(const float* __restrict__ Hin, const float* __restrict__ Wt,
                 const float* __restrict__ tb, const float* __restrict__ gam,
                 const float* __restrict__ bet, float* __restrict__ Out) {
  __shared__ alignas(16) _Float16 lsA[128][40];
  __shared__ alignas(16) _Float16 lsB[128][40];
  const int tid  = threadIdx.x;
  const int wid  = tid >> 5;
  const int lane = tid & 31;
  const int m0   = blockIdx.x * 128;

  v8f zero = {};
  v8f acc[8];
#pragma unroll
  for (int j = 0; j < 8; ++j) acc[j] = zero;

  for (int tap = 0; tap < 3; ++tap) {
    for (int k0 = 0; k0 < 128; k0 += 32) {
      // A: temporally shifted input, masked at block-of-4 boundaries
      {
        const int r  = tid >> 1;
        const int cb = (tid & 1) * 16;
        const int n  = m0 + r;
        const int ts = (n & 3) + tap - 1;
        const bool valid = (ts >= 0) && (ts <= 3);
        const float* ap = Hin + (size_t)(n + tap - 1) * 128 + k0 + cb;
        unsigned* dp = (unsigned*)&lsA[r][cb];
#pragma unroll
        for (int p = 0; p < 8; ++p) {
          const float v0 = valid ? ap[2 * p] : 0.f;
          const float v1 = valid ? ap[2 * p + 1] : 0.f;
          dp[p] = pack2h(v0, v1);
        }
      }
      // B: lsB[c][ci] = w[c, ci, tap], ci-pairs packed
      {
        const int kp = tid >> 4;
        const int c0 = (tid & 15) * 8;
#pragma unroll
        for (int i = 0; i < 8; ++i) {
          const int c = c0 + i;
          const float w0 = Wt[(size_t)c * 384 + (k0 + 2 * kp) * 3 + tap];
          const float w1 = Wt[(size_t)c * 384 + (k0 + 2 * kp + 1) * 3 + tap];
          *(unsigned*)&lsB[c][2 * kp] = pack2h(w0, w1);
        }
      }
      __syncthreads();

      const int ar  = wid * 16 + (lane & 15);
      const int akh = (lane >> 4) * 8;
      union { v16h v; v8h h[2]; } ua;
      ua.h[0] = *(const v8h*)&lsA[ar][akh];
      ua.h[1] = *(const v8h*)&lsA[ar][akh + 16];
#pragma unroll
      for (int j = 0; j < 8; ++j) {
        const int bn  = j * 16 + (lane & 15);
        const int bkh = (lane >> 4) * 8;
        union { v16h v; v8h h[2]; } ub;
        ub.h[0] = *(const v8h*)&lsB[bn][bkh];
        ub.h[1] = *(const v8h*)&lsB[bn][bkh + 16];
        acc[j] = __builtin_amdgcn_wmma_f32_16x16x32_f16(
            false, ua.v, false, ub.v, (short)0, acc[j], false, false);
      }
      __syncthreads();
    }
  }

  // ---- fused bias + LayerNorm over the 128 channels of each row ----
  const int hi   = lane >> 4;
  const int col0 = lane & 15;
  float bcol[8], gcol[8], bbet[8];
#pragma unroll
  for (int j = 0; j < 8; ++j) {
    const int c = j * 16 + col0;
    bcol[j] = tb[c]; gcol[j] = gam[c]; bbet[j] = bet[c];
  }
#pragma unroll
  for (int r = 0; r < 8; ++r) {
    float vv[8], s = 0.f, s2 = 0.f;
#pragma unroll
    for (int j = 0; j < 8; ++j) {
      const float t = acc[j][r] + bcol[j];
      vv[j] = t; s += t; s2 += t * t;
    }
    // reduce across the 16 lanes of this half-wave (one row per half)
    for (int m = 1; m < 16; m <<= 1) {
      s  += __shfl_xor(s,  m, 32);
      s2 += __shfl_xor(s2, m, 32);
    }
    const float mu  = s * (1.f / 128.f);
    const float var = s2 * (1.f / 128.f) - mu * mu;
    const float inv = rsqrtf(var + 1e-5f);
    const int row = m0 + wid * 16 + hi * 8 + r;
#pragma unroll
    for (int j = 0; j < 8; ++j)
      Out[(size_t)row * 128 + j * 16 + col0] = gcol[j] * (vv[j] - mu) * inv + bbet[j];
  }
}

// ---------------------------------------------------------------------------
extern "C" void kernel_launch(void* const* d_in, const int* in_sizes, int n_in,
                              void* d_out, int out_size, void* d_ws, size_t ws_size,
                              hipStream_t stream) {
  (void)n_in; (void)out_size; (void)ws_size;
  const float* x        = (const float*)d_in[0];
  const int*   ei       = (const int*)d_in[1];
  const float* W1       = (const float*)d_in[2];
  const float* att_src1 = (const float*)d_in[3];
  const float* att_dst1 = (const float*)d_in[4];
  const float* b1       = (const float*)d_in[5];
  const float* W2       = (const float*)d_in[6];
  const float* att_src2 = (const float*)d_in[7];
  const float* att_dst2 = (const float*)d_in[8];
  const float* b2       = (const float*)d_in[9];
  const float* Wr       = (const float*)d_in[10];
  const float* br       = (const float*)d_in[11];
  const float* tcn_w    = (const float*)d_in[12];
  const float* tcn_b    = (const float*)d_in[13];
  const float* gamma    = (const float*)d_in[14];
  const float* beta     = (const float*)d_in[15];
  float* out = (float*)d_out;

  const int N = in_sizes[0] / 64;   // 32768
  const int E = in_sizes[1] / 2;    // 1048576
  const int* src = ei;
  const int* dst = ei + E;

  // workspace layout (floats): 3 big [N*128] buffers + small attention arrays
  float* ws  = (float*)d_ws;
  const size_t big = (size_t)N * 128;
  float* h1  = ws;             // layer-1 x@W1 ; later reused as h2
  float* o1  = ws + big;       // layer-1 GAT output ; later reused as hfin
  float* res = ws + 2 * big;   // residual
  float* as1 = ws + 3 * big;
  float* ad1 = as1 + 2 * (size_t)N;
  float* z1  = ad1 + 2 * (size_t)N;
  float* as2 = z1  + 2 * (size_t)N;
  float* ad2 = as2 + (size_t)N;
  float* z2  = ad2 + (size_t)N;
  float* h2   = h1;  // alias (h1 dead after layer-1 edge_agg)
  float* hfin = o1;  // alias (o1 dead after GEMM2)

  const dim3 B256(256);
  const int gemmGrid  = N / 128;                // 256
  const int nodeGrid  = (N + 255) / 256;        // 128
  const int elemGrid  = (N * 128 + 255) / 256;  // 16384
  const int edgeGridT = (E + 255) / 256;        // 4096 (thread/edge)
  const int edgeGridW = (E + 7) / 8;            // 131072 (wave/edge)

  // ---- layer 1 dense ----
  gemm128_wmma<<<gemmGrid, B256, 0, stream>>>(x, W1, nullptr, h1, 64, 0);
  gemm128_wmma<<<gemmGrid, B256, 0, stream>>>(x, Wr, br, res, 64, 0);
  // ---- layer 1 attention softmax + aggregation ----
  attn_node<<<nodeGrid, B256, 0, stream>>>(h1, att_src1, att_dst1, as1, ad1, z1, N, 2);
  edge_denom<<<edgeGridT, B256, 0, stream>>>(src, dst, as1, ad1, z1, E, 2);
  agg_init<<<elemGrid, B256, 0, stream>>>(h1, b1, nullptr, as1, ad1, z1, o1, N, 2);
  edge_agg<<<edgeGridW, B256, 0, stream>>>(src, dst, h1, as1, ad1, z1, o1, E, 2);
  // ---- layer 2 dense (elu fused into A staging) ----
  gemm128_wmma<<<gemmGrid, B256, 0, stream>>>(o1, W2, nullptr, h2, 128, 1);
  // ---- layer 2 attention softmax + aggregation (+ b2 + residual) ----
  attn_node<<<nodeGrid, B256, 0, stream>>>(h2, att_src2, att_dst2, as2, ad2, z2, N, 1);
  edge_denom<<<edgeGridT, B256, 0, stream>>>(src, dst, as2, ad2, z2, E, 1);
  agg_init<<<elemGrid, B256, 0, stream>>>(h2, b2, res, as2, ad2, z2, hfin, N, 1);
  edge_agg<<<edgeGridW, B256, 0, stream>>>(src, dst, h2, as2, ad2, z2, hfin, E, 1);
  // ---- TCN + LayerNorm ----
  tcn_ln_wmma<<<gemmGrid, B256, 0, stream>>>(hfin, tcn_w, tcn_b, gamma, beta, out);
}
